// MultiHeadedAttention_46153718563352
// MI455X (gfx1250) — compile-verified, matrix operand order checked by reading, not run on hardware
//
#include <hip/hip_runtime.h>

typedef __attribute__((ext_vector_type(16))) _Float16 v16h;
typedef __attribute__((ext_vector_type(8)))  _Float16 v8h;
typedef __attribute__((ext_vector_type(8)))  float    v8f;

#define D_INQ  1024
#define HEADS  16
#define DK     64
#define BATCH  2
#define SEQ    2048

// ---- fragment load for NT GEMM operands (row-major, K contiguous) ----
// 16-bit A/B fragment layout (CDNA5 ISA 7.12.2): lane l holds row (l&15);
// lanes 0-15: K = {0..7, 16..23}; lanes 16-31: K = {8..15, 24..31}.
__device__ __forceinline__ v16h load_frag_nt(const _Float16* p, int ld) {
    const int lane = threadIdx.x & 31;
    const int r    = lane & 15;
    const int koff = (lane >> 4) << 3;          // 0 or 8
    const _Float16* row = p + (size_t)r * ld + koff;
    v8h lo = *(const v8h*)(row);                // K = koff .. koff+7
    v8h hi = *(const v8h*)(row + 16);           // K = koff+16 .. koff+23
    v16h f;
#pragma unroll
    for (int i = 0; i < 8; ++i) { f[i] = lo[i]; f[i + 8] = hi[i]; }
    return f;
}

__device__ __forceinline__ v8f wmma16(v16h a, v16h b, v8f c) {
    return __builtin_amdgcn_wmma_f32_16x16x32_f16(false, a, false, b,
                                                  (short)0, c, false, false);
}

__device__ __forceinline__ float rmax16(float v) {
    v = fmaxf(v, __shfl_xor(v, 1));
    v = fmaxf(v, __shfl_xor(v, 2));
    v = fmaxf(v, __shfl_xor(v, 4));
    v = fmaxf(v, __shfl_xor(v, 8));
    return v;
}
__device__ __forceinline__ float rsum16(float v) {
    v += __shfl_xor(v, 1);
    v += __shfl_xor(v, 2);
    v += __shfl_xor(v, 4);
    v += __shfl_xor(v, 8);
    return v;
}

// ---- fp32 -> f16 conversion ----
__global__ void mha_cvt_f16_kernel(const float* __restrict__ src,
                                   _Float16* __restrict__ dst, int n) {
    int i = blockIdx.x * blockDim.x + threadIdx.x;
    int stride = gridDim.x * blockDim.x;
    for (; i < n; i += stride) dst[i] = (_Float16)src[i];
}

// ---- NT GEMM: C[M,1024] = A[M,1024] @ B[1024,1024]^T + bias ----
// Wave tile: 32 rows x 64 cols (2 A-frags share 4 B-frags -> 8 WMMA/K-step).
// Double-buffered, K-loop unrolled x2: stage s computes while stage 1-s loads.
// No rotation copies: each buffer written only by loads, read only by WMMAs.
// mode 0: store f16 [b,h,s,d] (scaled)   mode 2: store f16 [b,h,d,s]
// mode 3: store f32 row-major [M,1024]
__global__ void __launch_bounds__(256)
mha_gemm_nt_kernel(const _Float16* __restrict__ A,
                   const _Float16* __restrict__ Bm,
                   const float* __restrict__ bias,
                   void* __restrict__ outp, int mode, float scale) {
    const int wave = threadIdx.x >> 5;
    const int lane = threadIdx.x & 31;
    const int col  = lane & 15;
    const int hi   = lane >> 4;

    const int m0    = blockIdx.x << 5;                  // 32 rows per wave
    const int nbase = ((blockIdx.y << 3) + wave) << 6;  // 64 cols per wave

    v8f acc[2][4];
#pragma unroll
    for (int t = 0; t < 4; ++t) {
        float bv = bias[nbase + t * 16 + col];
#pragma unroll
        for (int i = 0; i < 8; ++i) { acc[0][t][i] = bv; acc[1][t][i] = bv; }
    }

    const _Float16* A0 = A + (size_t)m0 * D_INQ;
    const _Float16* A1 = A0 + (size_t)16 * D_INQ;
    const _Float16* Bw = Bm + (size_t)nbase * D_INQ;

    v16h a0[2], a1[2], bf[2][4];

    // prologue: stage 0 <- k = 0
    a0[0] = load_frag_nt(A0, D_INQ);
    a1[0] = load_frag_nt(A1, D_INQ);
#pragma unroll
    for (int t = 0; t < 4; ++t)
        bf[0][t] = load_frag_nt(Bw + (size_t)(t * 16) * D_INQ, D_INQ);

#pragma unroll 1
    for (int k = 0; k < D_INQ; k += 64) {
        // stage 1 <- k+32 (overlaps stage-0 WMMAs)
        const int k1 = k + 32;
        a0[1] = load_frag_nt(A0 + k1, D_INQ);
        a1[1] = load_frag_nt(A1 + k1, D_INQ);
#pragma unroll
        for (int t = 0; t < 4; ++t)
            bf[1][t] = load_frag_nt(Bw + (size_t)(t * 16) * D_INQ + k1, D_INQ);

#pragma unroll
        for (int t = 0; t < 4; ++t) {
            acc[0][t] = wmma16(a0[0], bf[0][t], acc[0][t]);
            acc[1][t] = wmma16(a1[0], bf[0][t], acc[1][t]);
        }

        // stage 0 <- k+64 (wraps to 0 on last iter; overlaps stage-1 WMMAs)
        const int k2 = (k + 64 < D_INQ) ? k + 64 : 0;
        a0[0] = load_frag_nt(A0 + k2, D_INQ);
        a1[0] = load_frag_nt(A1 + k2, D_INQ);
#pragma unroll
        for (int t = 0; t < 4; ++t)
            bf[0][t] = load_frag_nt(Bw + (size_t)(t * 16) * D_INQ + k2, D_INQ);

#pragma unroll
        for (int t = 0; t < 4; ++t) {
            acc[0][t] = wmma16(a0[1], bf[1][t], acc[0][t]);
            acc[1][t] = wmma16(a1[1], bf[1][t], acc[1][t]);
        }
    }

    if (mode == 3) {
        float* out = (float*)outp;
#pragma unroll
        for (int mh = 0; mh < 2; ++mh)
#pragma unroll
            for (int i = 0; i < 8; ++i) {
                int tg = m0 + mh * 16 + i + 8 * hi;
#pragma unroll
                for (int t = 0; t < 4; ++t)
                    out[(size_t)tg * D_INQ + nbase + t * 16 + col] = acc[mh][t][i];
            }
    } else {
        _Float16* out = (_Float16*)outp;
#pragma unroll
        for (int mh = 0; mh < 2; ++mh)
#pragma unroll
            for (int i = 0; i < 8; ++i) {
                int tg = m0 + mh * 16 + i + 8 * hi;
                int b = tg >> 11;         // / SEQ
                int s = tg & (SEQ - 1);
#pragma unroll
                for (int t = 0; t < 4; ++t) {
                    int o = nbase + t * 16 + col;
                    int h = o >> 6, d = o & 63;
                    _Float16 v = (_Float16)(acc[mh][t][i] * scale);
                    if (mode == 2)   // V transposed: [b,h,d,s]
                        out[((size_t)(b * HEADS + h) * DK + d) * SEQ + s] = v;
                    else             // Q/K: [b,h,s,d]
                        out[((size_t)(b * HEADS + h) * SEQ + s) * DK + d] = v;
                }
            }
    }
}

// ---- fused flash attention: one wave per 16-row q tile per (b,h) ----
__global__ void __launch_bounds__(128)
mha_attn_kernel(const _Float16* __restrict__ Q,
                const _Float16* __restrict__ Km,
                const _Float16* __restrict__ Vt,
                const int* __restrict__ mask,
                _Float16* __restrict__ ctx) {
    __shared__ _Float16 lds_p[4][16 * 32];
    const int wave = threadIdx.x >> 5;
    const int lane = threadIdx.x & 31;
    const int col  = lane & 15;
    const int hi   = lane >> 4;

    int tile = blockIdx.x * 4 + wave;       // 0 .. B*H*(S/16)-1
    int b  = tile >> 11;
    int r0 = tile & 2047;
    int h  = r0 >> 7;
    int q0 = (r0 & 127) << 4;

    const _Float16* Qbh = Q  + (size_t)(b * HEADS + h) * SEQ * DK;
    const _Float16* Kbh = Km + (size_t)(b * HEADS + h) * SEQ * DK;
    const _Float16* Vbh = Vt + (size_t)(b * HEADS + h) * DK * SEQ;
    const int* mb = mask + (size_t)b * SEQ * SEQ;
    _Float16* lp = lds_p[wave];

    // Q already scaled by 1/sqrt(dk) at projection time
    v16h aq0 = load_frag_nt(Qbh + (size_t)q0 * DK,      DK);
    v16h aq1 = load_frag_nt(Qbh + (size_t)q0 * DK + 32, DK);

    float mrun[8], lrun[8];
#pragma unroll
    for (int i = 0; i < 8; ++i) { mrun[i] = -3.0e38f; lrun[i] = 0.f; }
    v8f zero = {};
    v8f acc[4];
#pragma unroll
    for (int t = 0; t < 4; ++t) acc[t] = zero;

    // pipeline prologue: K fragments for k0 = 0
    v16h kf[4];
    kf[0] = load_frag_nt(Kbh,            DK);
    kf[1] = load_frag_nt(Kbh + 32,       DK);
    kf[2] = load_frag_nt(Kbh + 16 * DK,      DK);
    kf[3] = load_frag_nt(Kbh + 16 * DK + 32, DK);

    for (int k0 = 0; k0 < SEQ; k0 += 32) {
        // V fragments for this step: independent of scores, load early
        v16h vF[4];
#pragma unroll
        for (int t = 0; t < 4; ++t)
            vF[t] = load_frag_nt(Vbh + (size_t)(t * 16) * SEQ + k0, SEQ);

        // scores: two 16x16 tiles (cols k0..k0+15, k0+16..k0+31), K-dim = 64
        v8f sA = zero, sB = zero;
        sA = wmma16(aq0, kf[0], sA);
        sA = wmma16(aq1, kf[1], sA);
        sB = wmma16(aq0, kf[2], sB);
        sB = wmma16(aq1, kf[3], sB);

        // preload next step's K fragments; overlaps with softmax VALU below
        const int k0n = (k0 + 32 < SEQ) ? k0 + 32 : 0;
        __builtin_prefetch(Kbh + (size_t)k0n * DK, 0, 3);
        __builtin_prefetch(Vbh + k0n, 0, 3);
        kf[0] = load_frag_nt(Kbh + (size_t)k0n * DK,            DK);
        kf[1] = load_frag_nt(Kbh + (size_t)k0n * DK + 32,       DK);
        kf[2] = load_frag_nt(Kbh + (size_t)(k0n + 16) * DK,      DK);
        kf[3] = load_frag_nt(Kbh + (size_t)(k0n + 16) * DK + 32, DK);

        // mask + online softmax (row r = i + 8*hi, col = lane&15)
#pragma unroll
        for (int i = 0; i < 8; ++i) {
            int r = i + 8 * hi;
            const int* mrow = mb + (size_t)(q0 + r) * SEQ + k0 + col;
            float sa = (mrow[0]  == 0) ? -1.0e9f : sA[i];
            float sb = (mrow[16] == 0) ? -1.0e9f : sB[i];
            float rm = rmax16(fmaxf(sa, sb));
            float mn = fmaxf(mrun[i], rm);
            float alpha = __expf(mrun[i] - mn);
            float pa = __expf(sa - mn);
            float pb = __expf(sb - mn);
            lrun[i] = lrun[i] * alpha + rsum16(pa + pb);
            mrun[i] = mn;
            lp[r * 32 + col]      = (_Float16)pa;   // stage P for A-frag reload
            lp[r * 32 + 16 + col] = (_Float16)pb;
#pragma unroll
            for (int t = 0; t < 4; ++t) acc[t][i] *= alpha;
        }

        // context += P(16x32) @ V(32x64): B frags from transposed V
        v16h pF = load_frag_nt(lp, 32);
#pragma unroll
        for (int t = 0; t < 4; ++t)
            acc[t] = wmma16(pF, vF[t], acc[t]);
    }

    // normalize and store context back as [B,S,1024] f16 for the output GEMM
#pragma unroll
    for (int i = 0; i < 8; ++i) {
        int r = i + 8 * hi;
        float inv = 1.0f / lrun[i];
        size_t base = (size_t)(b * SEQ + q0 + r) * D_INQ + h * DK;
#pragma unroll
        for (int t = 0; t < 4; ++t)
            ctx[base + t * 16 + col] = (_Float16)(acc[t][i] * inv);
    }
}

extern "C" void kernel_launch(void* const* d_in, const int* in_sizes, int n_in,
                              void* d_out, int out_size, void* d_ws, size_t ws_size,
                              hipStream_t stream) {
    (void)in_sizes; (void)n_in; (void)out_size; (void)ws_size;
    const float* x   = (const float*)d_in[0];
    const int*   msk = (const int*)d_in[1];
    const float* Wq  = (const float*)d_in[2];
    const float* bq  = (const float*)d_in[3];
    const float* Wk  = (const float*)d_in[4];
    const float* bk  = (const float*)d_in[5];
    const float* Wv  = (const float*)d_in[6];
    const float* bv  = (const float*)d_in[7];
    const float* Wo  = (const float*)d_in[8];
    const float* bo  = (const float*)d_in[9];
    float* out = (float*)d_out;

    const size_t NX = (size_t)BATCH * SEQ * D_INQ;   // 4M elems
    const size_t NW = (size_t)D_INQ * D_INQ;         // 1M elems
    char* ws = (char*)d_ws;
    _Float16* xh  = (_Float16*)(ws);
    _Float16* wqh = (_Float16*)(ws + NX * 2);
    _Float16* wkh = (_Float16*)(ws + NX * 2 + NW * 2);
    _Float16* wvh = (_Float16*)(ws + NX * 2 + NW * 4);
    _Float16* woh = (_Float16*)(ws + NX * 2 + NW * 6);
    _Float16* Qh  = (_Float16*)(ws + NX * 2 + NW * 8);
    _Float16* Kh  = (_Float16*)(ws + NX * 4 + NW * 8);
    _Float16* Vth = (_Float16*)(ws + NX * 6 + NW * 8);
    _Float16* Ch  = (_Float16*)(ws + NX * 8 + NW * 8);

    mha_cvt_f16_kernel<<<2048, 256, 0, stream>>>(x,  xh,  (int)NX);
    mha_cvt_f16_kernel<<<512,  256, 0, stream>>>(Wq, wqh, (int)NW);
    mha_cvt_f16_kernel<<<512,  256, 0, stream>>>(Wk, wkh, (int)NW);
    mha_cvt_f16_kernel<<<512,  256, 0, stream>>>(Wv, wvh, (int)NW);
    mha_cvt_f16_kernel<<<512,  256, 0, stream>>>(Wo, woh, (int)NW);

    dim3 gg(BATCH * SEQ / 32, 2);   // 128 M-tiles x (2 x 8 waves x 64 cols)
    const float qscale = 0.125f;    // 1/sqrt(DK)
    mha_gemm_nt_kernel<<<gg, 256, 0, stream>>>(xh, wqh, bq, (void*)Qh,  0, qscale);
    mha_gemm_nt_kernel<<<gg, 256, 0, stream>>>(xh, wkh, bk, (void*)Kh,  0, 1.0f);
    mha_gemm_nt_kernel<<<gg, 256, 0, stream>>>(xh, wvh, bv, (void*)Vth, 2, 1.0f);

    mha_attn_kernel<<<BATCH * HEADS * (SEQ / 16) / 4, 128, 0, stream>>>(
        Qh, Kh, Vth, msk, Ch);

    mha_gemm_nt_kernel<<<gg, 256, 0, stream>>>(Ch, woh, bo, (void*)out, 3, 1.0f);
}
